// Attention_49838800503122
// MI455X (gfx1250) — matrix operand order repaired and checked by reading, not compiled, not run on hardware
//
#include <hip/hip_runtime.h>
#include <hip/hip_bf16.h>

// ---------------- problem constants ----------------
#define BATCH   2
#define SEQ     4096
#define DIMX    768
#define NHEAD   12
#define HD      64
#define MBH     (BATCH * NHEAD)      // 24
#define BLK     32
#define NB      128                  // SEQ/BLK blocks per sequence
#define NSEL    512                  // selected blocks per (batch*head)
#define D3      (3 * DIMX)           // 2304 (Q|K|V packed output cols)
#define NTOK    (BATCH * SEQ)        // 8192

typedef __attribute__((ext_vector_type(16))) _Float16 v16h;
typedef __attribute__((ext_vector_type(8)))  float    v8f;

// ---------------- WMMA fragment loaders (contiguous k = 16*(lane/16) + s per lane) ----------------
// A matrix 16x32 f16, row-major source with leading dim lda (elements).
__device__ inline v16h load_a_frag(const _Float16* A, int lda, int row0, int k0) {
    int lane = threadIdx.x & 31;
    const _Float16* p = A + (size_t)(row0 + (lane & 15)) * lda + k0;
    int kh = (lane >> 4) & 1;
    v16h a;
#pragma unroll
    for (int v = 0; v < 8; ++v) {
        int kk = kh * 16 + 2 * v;
        a[2 * v]     = p[kk];
        a[2 * v + 1] = p[kk + 1];
    }
    return a;
}

// B matrix 32x16 f16 (KxN), row-major source with leading dim ldb.
__device__ inline v16h load_b_frag(const _Float16* Bm, int ldb, int k0, int col0) {
    int lane = threadIdx.x & 31;
    int c  = col0 + (lane & 15);
    int kg = (lane >> 4) & 1;
    v16h b;
#pragma unroll
    for (int v = 0; v < 8; ++v) {
        int kk = k0 + kg * 16 + 2 * v;
        b[2 * v]     = Bm[(size_t)kk * ldb + c];
        b[2 * v + 1] = Bm[(size_t)(kk + 1) * ldb + c];
    }
    return b;
}

// B = src^T where src is row-major [N][K]; B(k,n) = src[n][k].
// Per lane: 16 contiguous halves -> two global_load_b128.
__device__ inline v16h load_bT_frag(const _Float16* src, int ld, int n0, int k0) {
    int lane = threadIdx.x & 31;
    const _Float16* p = src + (size_t)(n0 + (lane & 15)) * ld + k0 + ((lane >> 4) & 1) * 16;
    v16h b;
#pragma unroll
    for (int v = 0; v < 8; ++v) {
        b[2 * v]     = p[2 * v];
        b[2 * v + 1] = p[2 * v + 1];
    }
    return b;
}

__device__ inline void atomicMaxFloat(float* addr, float val) {
    unsigned* ua  = (unsigned*)addr;
    unsigned  old = *ua;
    while (__uint_as_float(old) < val) {
        unsigned assumed = old;
        old = atomicCAS(ua, assumed, __float_as_uint(val));
        if (old == assumed) break;
    }
}

// monotonic float -> sortable uint
__device__ inline unsigned f2s(float f) {
    unsigned b = __float_as_uint(f);
    return (b & 0x80000000u) ? ~b : (b | 0x80000000u);
}

// ---------------- conversion / packing ----------------
__global__ void k_cvt_f16(const float* __restrict__ src, _Float16* __restrict__ dst, int n) {
    int i = blockIdx.x * blockDim.x + threadIdx.x;
    if (i < n) dst[i] = (_Float16)src[i];
}

// Pack Wq|Wk|Wv transposed: Wt[(which*DIMX + j)][k] = W[k][j]  (K contiguous per row)
__global__ void k_pack_qkvw(const float* __restrict__ Wq, const float* __restrict__ Wk,
                            const float* __restrict__ Wv, const float* __restrict__ bq,
                            const float* __restrict__ bk, const float* __restrict__ bv,
                            _Float16* __restrict__ Wt, float* __restrict__ bc) {
    int i = blockIdx.x * blockDim.x + threadIdx.x;
    if (i < DIMX * DIMX) {
        int k = i / DIMX, j = i % DIMX;
        Wt[(size_t)j * DIMX + k]              = (_Float16)Wq[i];
        Wt[(size_t)(j + DIMX) * DIMX + k]     = (_Float16)Wk[i];
        Wt[(size_t)(j + 2 * DIMX) * DIMX + k] = (_Float16)Wv[i];
    }
    if (i < DIMX) { bc[i] = bq[i]; bc[i + DIMX] = bk[i]; bc[i + 2 * DIMX] = bv[i]; }
}

// Transpose + convert: dst[j][k] = src[k][j], src row-major [K=768][N=768]
__global__ void k_cvt_f16_t(const float* __restrict__ src, _Float16* __restrict__ dst) {
    int i = blockIdx.x * blockDim.x + threadIdx.x;
    if (i < DIMX * DIMX) {
        int k = i / DIMX, j = i % DIMX;
        dst[(size_t)j * DIMX + k] = (_Float16)src[i];
    }
}

// ---------------- fused QKV projection: [8192,768] x [768,2304] ----------------
// B supplied transposed [D3][DIMX]. Each wave computes a 16x64 strip (4 N-tiles).
__global__ __launch_bounds__(128) void k_gemm_qkv(
    const _Float16* __restrict__ Xh, const _Float16* __restrict__ Wt,
    const float* __restrict__ bias, const float* __restrict__ mask,
    _Float16* __restrict__ Qo, _Float16* __restrict__ Ko, _Float16* __restrict__ Vo) {
    const int NG = (D3 / 16) / 4;            // 36 N-groups of 4 tiles
    int wave = threadIdx.x >> 5, lane = threadIdx.x & 31;
    int wid = blockIdx.x * 4 + wave;
    int mt = wid / NG, ng = wid % NG;
    int nt0 = ng * 4;
    v8f acc[4] = {{}, {}, {}, {}};
    for (int k0 = 0; k0 < DIMX; k0 += 32) {
        // hoist all fragment loads so VMEM latency overlaps the WMMA sequence
        v16h a = load_a_frag(Xh, DIMX, mt * 16, k0);
        v16h bfr[4];
#pragma unroll
        for (int j = 0; j < 4; ++j)
            bfr[j] = load_bT_frag(Wt, DIMX, (nt0 + j) * 16, k0);
        if (k0 + 32 < DIMX) {
            __builtin_prefetch(Xh + (size_t)(mt * 16 + (lane & 15)) * DIMX + k0 + 32, 0, 3);
            __builtin_prefetch(Wt + (size_t)(nt0 * 16 + (lane & 15)) * DIMX + k0 + 32, 0, 3);
        }
#pragma unroll
        for (int j = 0; j < 4; ++j)
            acc[j] = __builtin_amdgcn_wmma_f32_16x16x32_f16(false, a, false, bfr[j], (short)0,
                                                            acc[j], false, false);
    }
    int rowadd = (lane >> 4) << 3;
#pragma unroll
    for (int j = 0; j < 4; ++j) {
        int col = (nt0 + j) * 16 + (lane & 15);
        float bval = bias[col];
        int which = col / DIMX;
        int w = col % DIMX;
        int head = w >> 6, dim = w & 63;
#pragma unroll
        for (int vr = 0; vr < 8; ++vr) {
            int row = mt * 16 + vr + rowadd;      // global token 0..8191
            float m   = mask[row];
            float val = (acc[j][vr] + bval) * m;
            int bb = row >> 12, s = row & (SEQ - 1);
            size_t di = (((size_t)(bb * NHEAD + head) << 12) + s) * HD + dim;
            _Float16 hv = (_Float16)val;
            if (which == 0) Qo[di] = hv;
            else if (which == 1) Ko[di] = hv;
            else Vo[di] = hv;
        }
    }
}

// ---------------- block means ----------------
__global__ void k_block_means(const _Float16* __restrict__ Q, const _Float16* __restrict__ K,
                              const _Float16* __restrict__ V, const float* __restrict__ mask,
                              float* __restrict__ Qm, float* __restrict__ Km,
                              float* __restrict__ Vm, float* __restrict__ tc) {
    int mbn = blockIdx.x;                 // 0..MBH*NB-1
    int mb = mbn >> 7, n = mbn & (NB - 1);
    int d = threadIdx.x;                  // 0..63
    int b = mb / NHEAD;
    size_t base = (((size_t)mb << 12) + n * BLK) * HD;
    float sq = 0.f, sk = 0.f, sv = 0.f;
    for (int r = 0; r < BLK; ++r) {
        size_t i = base + (size_t)r * HD + d;
        sq += (float)Q[i]; sk += (float)K[i]; sv += (float)V[i];
    }
    float t = 0.f;
    for (int r = 0; r < BLK; ++r) t += mask[b * SEQ + n * BLK + r];
    float denom = t + 1e-6f;
    size_t o = ((size_t)mb * NB + n) * HD + d;
    Qm[o] = sq / denom; Km[o] = sk / denom; Vm[o] = sv / denom;
    if (d == 0) tc[mb * NB + n] = t;
}

// ---------------- low-res logits + row max + selection scores ----------------
__global__ __launch_bounds__(128) void k_lowlogit(
    const float* __restrict__ Qm, const float* __restrict__ Km, const float* __restrict__ tc,
    float* __restrict__ low_logit, float* __restrict__ row_max, float* __restrict__ sel) {
    __shared__ float qrow[HD];
    __shared__ float red[128];
    int mbn = blockIdx.x;
    int mb = mbn >> 7, n = mbn & (NB - 1);
    int m = threadIdx.x;                  // 0..127 key block
    if (m < HD) qrow[m] = Qm[((size_t)mb * NB + n) * HD + m];
    __syncthreads();
    const float* kr = Km + ((size_t)mb * NB + m) * HD;
    float dot = 0.f;
    for (int d = 0; d < HD; ++d) dot += qrow[d] * kr[d];
    dot *= 0.125f;                        // 1/sqrt(64)
    red[m] = dot;
    __syncthreads();
    for (int off = 64; off; off >>= 1) {
        if (m < off) red[m] = fmaxf(red[m], red[m + off]);
        __syncthreads();
    }
    float rmax = red[0];
    float tcn = tc[mb * NB + n], tcm = tc[mb * NB + m];
    float empty = (tcm * tcn < 0.5f) ? 1.f : 0.f;
    float lo = dot - 1e4f * empty;
    size_t o = (size_t)mbn * NB + m;
    low_logit[o] = lo;
    float s = lo - rmax;
    int diff = n - m; if (diff < 0) diff = -diff;
    if (diff <= 1) s += 5e3f;             // DIAG_N=3 -> off=1
    if (n < 1) s += 5e3f;                 // FIRST_N=1
    if (m < 1) s += 5e3f;
    sel[o] = s;
    if (m == 0) row_max[mbn] = rmax;
}

// ---------------- exact top-512 per mb (bitwise threshold search) ----------------
__global__ void k_topk(const float* __restrict__ sel, float* __restrict__ hr_mask,
                       int* __restrict__ idx) {
    __shared__ unsigned s_cnt;
    int mb = blockIdx.x;
    const float* p = sel + (size_t)mb * NB * NB;
    int tid = threadIdx.x, nth = blockDim.x;
    unsigned thr = 0;
    for (int bit = 31; bit >= 0; --bit) {
        unsigned cand = thr | (1u << bit);
        if (tid == 0) s_cnt = 0;
        __syncthreads();
        unsigned c = 0;
        for (int i = tid; i < NB * NB; i += nth)
            if (f2s(p[i]) >= cand) ++c;
        atomicAdd(&s_cnt, c);
        __syncthreads();
        if (s_cnt >= NSEL) thr = cand;
        __syncthreads();
    }
    for (int i = tid; i < NB * NB; i += nth)
        hr_mask[(size_t)mb * NB * NB + i] = (f2s(p[i]) >= thr) ? 1.f : 0.f;
    if (tid == 0) s_cnt = 0;
    __syncthreads();
    unsigned cg = 0;
    for (int i = tid; i < NB * NB; i += nth)
        if (f2s(p[i]) > thr) ++cg;
    atomicAdd(&s_cnt, cg);
    __syncthreads();
    if (tid == 0) {
        int need_eq = NSEL - (int)s_cnt;
        int pos = 0;
        for (int i = 0; i < NB * NB && pos < NSEL; ++i) {
            unsigned u = f2s(p[i]);
            if (u > thr) idx[mb * NSEL + pos++] = i;
            else if (u == thr && need_eq > 0) { idx[mb * NSEL + pos++] = i; --need_eq; }
        }
    }
}

// ---------------- init accumulators ----------------
__global__ void k_init(float* __restrict__ hr_out, float* __restrict__ hr_norm,
                       float* __restrict__ max_vals) {
    int i = blockIdx.x * blockDim.x + threadIdx.x;
    if (i < MBH * SEQ * HD) hr_out[i] = 0.f;
    if (i < MBH * SEQ) { hr_norm[i] = 0.f; max_vals[i] = -1e9f; }
}

// ---------------- HR pass 1: block QK^T, scatter row max ----------------
__global__ __launch_bounds__(128) void k_hr_pass1(
    const _Float16* __restrict__ Q, const _Float16* __restrict__ K,
    const int* __restrict__ idx, float* __restrict__ max_vals) {
    __shared__ float s_logit[BLK][BLK + 1];
    int gb = blockIdx.x;                  // 0..MBH*NSEL-1
    int mb = gb / NSEL;
    int e  = idx[gb];
    int qb = e >> 7, kb = e & (NB - 1);
    const _Float16* Qb = Q + ((size_t)mb << 12) * HD;
    const _Float16* Kb = K + ((size_t)mb << 12) * HD;
    int wave = threadIdx.x >> 5, lane = threadIdx.x & 31;
    int mi = wave >> 1, ni = wave & 1;
    v16h a0 = load_a_frag(Qb, HD, qb * BLK + mi * 16, 0);
    v16h a1 = load_a_frag(Qb, HD, qb * BLK + mi * 16, 32);
    v16h b0 = load_bT_frag(Kb, HD, kb * BLK + ni * 16, 0);
    v16h b1 = load_bT_frag(Kb, HD, kb * BLK + ni * 16, 32);
    v8f acc = {};
    acc = __builtin_amdgcn_wmma_f32_16x16x32_f16(false, a0, false, b0, (short)0, acc,
                                                 false, false);
    acc = __builtin_amdgcn_wmma_f32_16x16x32_f16(false, a1, false, b1, (short)0, acc,
                                                 false, false);
    int rowa = (lane >> 4) << 3;
    int col  = ni * 16 + (lane & 15);
#pragma unroll
    for (int vr = 0; vr < 8; ++vr)
        s_logit[mi * 16 + vr + rowa][col] = acc[vr] * 0.125f;
    __syncthreads();
    if (threadIdx.x < BLK) {
        int r = threadIdx.x;
        float mx = s_logit[r][0];
        for (int c = 1; c < BLK; ++c) mx = fmaxf(mx, s_logit[r][c]);
        atomicMaxFloat(&max_vals[(mb << 12) + qb * BLK + r], mx);
    }
}

// ---------------- HR pass 2: exp, scatter-add out & norm ----------------
__global__ __launch_bounds__(128) void k_hr_pass2(
    const _Float16* __restrict__ Q, const _Float16* __restrict__ K,
    const _Float16* __restrict__ V, const int* __restrict__ idx,
    const float* __restrict__ mask, const float* __restrict__ max_vals,
    float* __restrict__ hr_out, float* __restrict__ hr_norm) {
    __shared__ float    s_w[BLK][BLK + 1];
    __shared__ _Float16 s_a[BLK][BLK];
    int gb = blockIdx.x;
    int mb = gb / NSEL;
    int e  = idx[gb];
    int qb = e >> 7, kb = e & (NB - 1);
    int b  = mb / NHEAD;
    const _Float16* Qb = Q + ((size_t)mb << 12) * HD;
    const _Float16* Kb = K + ((size_t)mb << 12) * HD;
    int wave = threadIdx.x >> 5, lane = threadIdx.x & 31;
    {   // recompute scaled logits
        int mi = wave >> 1, ni = wave & 1;
        v16h a0 = load_a_frag(Qb, HD, qb * BLK + mi * 16, 0);
        v16h a1 = load_a_frag(Qb, HD, qb * BLK + mi * 16, 32);
        v16h b0 = load_bT_frag(Kb, HD, kb * BLK + ni * 16, 0);
        v16h b1 = load_bT_frag(Kb, HD, kb * BLK + ni * 16, 32);
        v8f acc = {};
        acc = __builtin_amdgcn_wmma_f32_16x16x32_f16(false, a0, false, b0, (short)0, acc,
                                                     false, false);
        acc = __builtin_amdgcn_wmma_f32_16x16x32_f16(false, a1, false, b1, (short)0, acc,
                                                     false, false);
        int rowa = (lane >> 4) << 3;
        int col  = ni * 16 + (lane & 15);
#pragma unroll
        for (int vr = 0; vr < 8; ++vr)
            s_w[mi * 16 + vr + rowa][col] = acc[vr] * 0.125f;
    }
    __syncthreads();
    for (int t = threadIdx.x; t < BLK * BLK; t += blockDim.x) {
        int r = t >> 5, c = t & 31;
        float km = mask[b * SEQ + kb * BLK + c];
        float wv = expf(s_w[r][c] - max_vals[(mb << 12) + qb * BLK + r] - 1e4f * (1.f - km));
        s_w[r][c] = wv;
        s_a[r][c] = (_Float16)wv;
    }
    __syncthreads();
    if (threadIdx.x < BLK) {
        int r = threadIdx.x;
        float ssum = 0.f;
        for (int c = 0; c < BLK; ++c) ssum += s_w[r][c];
        atomicAdd(&hr_norm[(mb << 12) + qb * BLK + r], ssum);
    }
    // attn(32x32) @ Vb(32x64) via 8 WMMA tiles, 2 per wave
    const _Float16* Vb = V + ((size_t)mb << 12) * HD + (size_t)(kb * BLK) * HD;
#pragma unroll
    for (int t2 = 0; t2 < 2; ++t2) {
        int tileid = wave + t2 * 4;
        int mi2 = tileid >> 2, nj = tileid & 3;
        v16h a = load_a_frag(&s_a[0][0], BLK, mi2 * 16, 0);
        v16h bf = load_b_frag(Vb, HD, 0, nj * 16);
        v8f o = {};
        o = __builtin_amdgcn_wmma_f32_16x16x32_f16(false, a, false, bf, (short)0, o,
                                                   false, false);
        int row0 = qb * BLK + mi2 * 16 + ((lane >> 4) << 3);
        int colo = nj * 16 + (lane & 15);
#pragma unroll
        for (int vr = 0; vr < 8; ++vr)
            atomicAdd(&hr_out[(((size_t)(mb << 12)) + row0 + vr) * HD + colo], o[vr]);
    }
}

// ---------------- LR path + merge, emit f16 attn in [B,S,DIM] ----------------
__global__ __launch_bounds__(128) void k_lr_merge(
    const float* __restrict__ low_logit, const float* __restrict__ row_max,
    const float* __restrict__ hr_mask, const float* __restrict__ tc,
    const float* __restrict__ Vm, const float* __restrict__ mask,
    const float* __restrict__ max_vals, const float* __restrict__ hr_out,
    const float* __restrict__ hr_norm, _Float16* __restrict__ attnh) {
    __shared__ float wsh[NB];
    __shared__ float lrout[HD];
    __shared__ float lrnorm_s;
    int mbn = blockIdx.x;
    int mb = mbn >> 7, n = mbn & (NB - 1);
    int b = mb / NHEAD, head = mb % NHEAD;
    int tid = threadIdx.x;
    float rmax = row_max[mbn];
    {
        int m = tid;
        float lo = low_logit[(size_t)mbn * NB + m];
        float hm = hr_mask[(size_t)mbn * NB + m];
        wsh[m] = expf(lo - rmax - 1e4f * hm) * tc[mb * NB + m];
    }
    __syncthreads();
    if (tid == 0) {
        float s = 0.f;
        for (int m = 0; m < NB; ++m) s += wsh[m];
        lrnorm_s = s;
    }
    if (tid < HD) {
        float s = 0.f;
        for (int m = 0; m < NB; ++m) s += wsh[m] * Vm[((size_t)mb * NB + m) * HD + tid];
        lrout[tid] = s;
    }
    __syncthreads();
    float lrn = lrnorm_s;
    for (int r = 0; r < BLK; ++r) {
        int s_ = n * BLK + r;
        float mk = mask[b * SEQ + s_];
        float mv = max_vals[(mb << 12) + s_];
        float lc = (rmax - mv) * mk;
        float lrc = expf(fminf(lc, 0.f));
        float hrc = expf(-fmaxf(lc, 0.f));
        float den = hr_norm[(mb << 12) + s_] * hrc + lrn * lrc + 1e-6f;
        if (tid < HD) {
            float ho = hr_out[(((size_t)(mb << 12)) + s_) * HD + tid];
            float o  = (ho * hrc + lrout[tid] * lrc) / den * mk;
            attnh[((size_t)(b * SEQ + s_)) * DIMX + head * HD + tid] = (_Float16)o;
        }
    }
}

// ---------------- output projection: [8192,768] x [768,768]^T -> d_out fp32 ----------------
__global__ __launch_bounds__(128) void k_gemm_out(
    const _Float16* __restrict__ Ah, const _Float16* __restrict__ Wt,
    const float* __restrict__ bo, float* __restrict__ out) {
    const int NG = (DIMX / 16) / 4;           // 12 N-groups of 4 tiles
    int wave = threadIdx.x >> 5, lane = threadIdx.x & 31;
    int wid = blockIdx.x * 4 + wave;
    int mt = wid / NG, ng = wid % NG;
    int nt0 = ng * 4;
    v8f acc[4] = {{}, {}, {}, {}};
    for (int k0 = 0; k0 < DIMX; k0 += 32) {
        v16h a = load_a_frag(Ah, DIMX, mt * 16, k0);
        v16h bfr[4];
#pragma unroll
        for (int j = 0; j < 4; ++j)
            bfr[j] = load_bT_frag(Wt, DIMX, (nt0 + j) * 16, k0);
        if (k0 + 32 < DIMX) {
            __builtin_prefetch(Ah + (size_t)(mt * 16 + (lane & 15)) * DIMX + k0 + 32, 0, 3);
            __builtin_prefetch(Wt + (size_t)(nt0 * 16 + (lane & 15)) * DIMX + k0 + 32, 0, 3);
        }
#pragma unroll
        for (int j = 0; j < 4; ++j)
            acc[j] = __builtin_amdgcn_wmma_f32_16x16x32_f16(false, a, false, bfr[j], (short)0,
                                                            acc[j], false, false);
    }
    int rowadd = (lane >> 4) << 3;
#pragma unroll
    for (int j = 0; j < 4; ++j) {
        int col = (nt0 + j) * 16 + (lane & 15);
        float bval = bo[col];
#pragma unroll
        for (int vr = 0; vr < 8; ++vr)
            out[(size_t)(mt * 16 + vr + rowadd) * DIMX + col] = acc[j][vr] + bval;
    }
}

// ---------------- launch ----------------
extern "C" void kernel_launch(void* const* d_in, const int* in_sizes, int n_in,
                              void* d_out, int out_size, void* d_ws, size_t ws_size,
                              hipStream_t stream) {
    const float* X    = (const float*)d_in[0];
    const float* mask = (const float*)d_in[1];
    const float* Wq   = (const float*)d_in[2];
    const float* bq   = (const float*)d_in[3];
    const float* Wk   = (const float*)d_in[4];
    const float* bk   = (const float*)d_in[5];
    const float* Wv   = (const float*)d_in[6];
    const float* bv   = (const float*)d_in[7];
    const float* Wo   = (const float*)d_in[8];
    const float* bo   = (const float*)d_in[9];
    float* out = (float*)d_out;

    char* wsb = (char*)d_ws;
    size_t off = 0;
    auto alloc = [&](size_t bytes) -> void* {
        void* p = wsb + off;
        off = (off + bytes + 255) & ~(size_t)255;
        return p;
    };
    _Float16* Xh    = (_Float16*)alloc((size_t)NTOK * DIMX * 2);
    _Float16* Wtq   = (_Float16*)alloc((size_t)D3 * DIMX * 2);   // transposed [D3][DIMX]
    float*    bqkv  = (float*)   alloc((size_t)D3 * 4);
    _Float16* Wot   = (_Float16*)alloc((size_t)DIMX * DIMX * 2); // transposed [DIMX][DIMX]
    _Float16* Q16   = (_Float16*)alloc((size_t)MBH * SEQ * HD * 2);
    _Float16* K16   = (_Float16*)alloc((size_t)MBH * SEQ * HD * 2);
    _Float16* V16   = (_Float16*)alloc((size_t)MBH * SEQ * HD * 2);
    float*    tc    = (float*)   alloc((size_t)MBH * NB * 4);
    float*    Qm    = (float*)   alloc((size_t)MBH * NB * HD * 4);
    float*    Km    = (float*)   alloc((size_t)MBH * NB * HD * 4);
    float*    Vm    = (float*)   alloc((size_t)MBH * NB * HD * 4);
    float*    lowl  = (float*)   alloc((size_t)MBH * NB * NB * 4);
    float*    rmaxb = (float*)   alloc((size_t)MBH * NB * 4);
    float*    selb  = (float*)   alloc((size_t)MBH * NB * NB * 4);
    float*    hrm   = (float*)   alloc((size_t)MBH * NB * NB * 4);
    int*      idxb  = (int*)     alloc((size_t)MBH * NSEL * 4);
    float*    maxv  = (float*)   alloc((size_t)MBH * SEQ * 4);
    float*    hrnrm = (float*)   alloc((size_t)MBH * SEQ * 4);
    float*    hrout = (float*)   alloc((size_t)MBH * SEQ * HD * 4);
    _Float16* attnh = (_Float16*)alloc((size_t)NTOK * DIMX * 2);
    (void)ws_size; (void)n_in; (void)in_sizes; (void)out_size;

    // 1) conversions / packing (weights transposed for contiguous-K B fragments)
    {
        int n = NTOK * DIMX;
        k_cvt_f16<<<(n + 255) / 256, 256, 0, stream>>>(X, Xh, n);
        int nw = DIMX * DIMX;
        k_cvt_f16_t<<<(nw + 255) / 256, 256, 0, stream>>>(Wo, Wot);
        k_pack_qkvw<<<(nw + 255) / 256, 256, 0, stream>>>(Wq, Wk, Wv, bq, bk, bv, Wtq, bqkv);
    }
    // 2) fused QKV projection (WMMA, 1x4 register blocking)
    {
        int waves = (NTOK / 16) * ((D3 / 16) / 4);   // 512 * 36
        k_gemm_qkv<<<waves / 4, 128, 0, stream>>>(Xh, Wtq, bqkv, mask, Q16, K16, V16);
    }
    // 3) block means + token counts
    k_block_means<<<MBH * NB, HD, 0, stream>>>(Q16, K16, V16, mask, Qm, Km, Vm, tc);
    // 4) low-res logits, row max, selection scores
    k_lowlogit<<<MBH * NB, 128, 0, stream>>>(Qm, Km, tc, lowl, rmaxb, selb);
    // 5) exact top-512 per mb
    k_topk<<<MBH, 256, 0, stream>>>(selb, hrm, idxb);
    // 6) init accumulators
    {
        int n = MBH * SEQ * HD;
        k_init<<<(n + 255) / 256, 256, 0, stream>>>(hrout, hrnrm, maxv);
    }
    // 7) HR pass 1: scatter row max (WMMA)
    k_hr_pass1<<<MBH * NSEL, 128, 0, stream>>>(Q16, K16, idxb, maxv);
    // 8) HR pass 2: exp + scatter-add (WMMA)
    k_hr_pass2<<<MBH * NSEL, 128, 0, stream>>>(Q16, K16, V16, idxb, mask, maxv, hrout, hrnrm);
    // 9) LR path + merge -> f16 attn [B,S,DIM]
    k_lr_merge<<<MBH * NB, 128, 0, stream>>>(lowl, rmaxb, hrm, tc, Vm, mask, maxv, hrout,
                                             hrnrm, attnh);
    // 10) output projection (WMMA, 1x4 register blocking) -> d_out fp32
    {
        int waves = (NTOK / 16) * ((DIMX / 16) / 4); // 512 * 12
        k_gemm_out<<<waves / 4, 128, 0, stream>>>(attnh, Wot, bo, out);
    }
}
